// PersistentMemory_87866440941860
// MI455X (gfx1250) — matrix operand order repaired and checked by reading, not compiled, not run on hardware
//
#include <hip/hip_runtime.h>

#define BH_   32
#define H_    16
#define M_    1024
#define D_    64
#define P_    1024
#define SPAN_ 1024

#define MTILE  64      // M rows per workgroup
#define STRIDE 1025    // LDS row stride in floats (bank shift 1 word/row)

typedef __attribute__((ext_vector_type(16))) __bf16 v16bf;
typedef __attribute__((ext_vector_type(8)))  float  v8f;

// f32 -> bf16 via native convert (backend should pack pairs with v_cvt_pk_bf16_f32)
static __device__ __forceinline__ __bf16 f2b(float x) {
    return (__bf16)x;
}

// 16-bit A-matrix 16x32 slot->K map (ISA 7.12.2): VGPR0..3 hold K 0..7 / 8..15,
// VGPR4..7 hold K 16..23 / 24..31 depending on lane half h = lane>>4.
static __device__ __forceinline__ int kA(int j, int h) {
    return ((j & 8) << 1) + (j & 7) + (h << 3);
}

__global__ __launch_bounds__(256)
void pers_mem_fused(const float* __restrict__ q,
                    const float* __restrict__ attn,
                    const float* __restrict__ pkey,
                    const float* __restrict__ pval,
                    float* __restrict__ attn_out,
                    float* __restrict__ out)
{
    extern __shared__ float smem[];
    float* S    = smem;                  // MTILE x STRIDE persistent-score tile
    float* invd = smem + MTILE * STRIDE; // per-row 1/denominator

    const int tid  = threadIdx.x;
    const int lane = tid & 31;
    const int w    = tid >> 5;           // wave id 0..7 (wave32)
    const int bh   = blockIdx.y;         // 0..31
    const int h    = bh & (H_ - 1);
    const int m0   = blockIdx.x * MTILE;

    const int ln = lane & 15;            // N / M index inside fragment
    const int lh = lane >> 4;            // lane half

    // ---------------- GEMM1: S[64x1024] = Q_tile[64x64] * K[64x1024] (bf16 WMMA)
    // A-fragments for all 4 row-tiles stay resident; each wave owns 8 p-tiles,
    // so each K element is fetched+converted exactly once per workgroup.
    {
        v16bf A[4][2];
        #pragma unroll
        for (int rt = 0; rt < 4; ++rt) {
            const float* qr = q + ((size_t)bh * M_ + (m0 + rt * 16 + ln)) * D_;
            #pragma unroll
            for (int j = 0; j < 16; ++j) {
                const int k = kA(j, lh);
                A[rt][0][j] = f2b(qr[k]);
                A[rt][1][j] = f2b(qr[k + 32]);
            }
        }
        const float* kb = pkey + (size_t)h * D_ * P_;
        for (int i = 0; i < 8; ++i) {
            const int p0 = (w * 8 + i) * 16;
            v16bf B0, B1;
            #pragma unroll
            for (int j = 0; j < 16; ++j) {   // B 32x16: slot j -> K = kA(j, lh), col = ln
                const int d0 = kA(j, lh);
                B0[j] = f2b(kb[(size_t)d0 * P_ + p0 + ln]);
                B1[j] = f2b(kb[(size_t)(d0 + 32) * P_ + p0 + ln]);
            }
            #pragma unroll
            for (int rt = 0; rt < 4; ++rt) {
                v8f c = {};
                c = __builtin_amdgcn_wmma_f32_16x16x32_bf16(false, A[rt][0], false, B0, (short)0, c, false, false);
                c = __builtin_amdgcn_wmma_f32_16x16x32_bf16(false, A[rt][1], false, B1, (short)0, c, false, false);
                float* sr = S + (size_t)(rt * 16 + lh * 8) * STRIDE + p0 + ln;
                #pragma unroll
                for (int r = 0; r < 8; ++r) sr[r * STRIDE] = c[r];  // C: m = r + 8*lh, n = ln
            }
        }
    }
    __syncthreads();

    // ---------------- Online softmax over [attn/8  |  S], 8 rows per wave
    for (int rr = 0; rr < 8; ++rr) {
        const int r = w * 8 + rr;
        const int m = m0 + r;
        const float* arow = attn + ((size_t)bh * M_ + m) * SPAN_;
        float* srow = S + (size_t)r * STRIDE;

        float mx = -3.402823466e38f, sm = 0.f;
        for (int c = lane; c < SPAN_; c += 32) {   // streamed logits (HBM read #1)
            const float v  = arow[c] * 0.125f;
            const float nm = fmaxf(mx, v);
            sm = sm * __expf(mx - nm) + __expf(v - nm);
            mx = nm;
        }
        for (int c = lane; c < P_; c += 32) {      // persistent logits (LDS)
            const float v  = srow[c];
            const float nm = fmaxf(mx, v);
            sm = sm * __expf(mx - nm) + __expf(v - nm);
            mx = nm;
        }
        #pragma unroll
        for (int off = 16; off > 0; off >>= 1) {   // wave32 (m,s) combine
            const float om = __shfl_xor(mx, off, 32);
            const float os = __shfl_xor(sm, off, 32);
            const float nm = fmaxf(mx, om);
            sm = sm * __expf(mx - nm) + os * __expf(om - nm);
            mx = nm;
        }
        const float inv = 1.f / sm;
        if (lane == 0) invd[r] = inv;

        for (int c = lane; c < P_; c += 32)        // S := un-normalized probs
            srow[c] = __expf(srow[c] - mx);

        float* orow = attn_out + ((size_t)bh * M_ + m) * SPAN_;
        for (int c = lane; c < SPAN_; c += 32) {   // HBM read #2 (L2-hot) + NT write
            const float val = __expf(arow[c] * 0.125f - mx) * inv;
            __builtin_nontemporal_store(val, orow + c);
        }
    }
    __syncthreads();

    // ---------------- GEMM2: out[64x64] = probs_pers[64x1024] * V[1024x64]
    // Wave owns one 16-col d-tile (ct) and two 16-row tiles: V fragment loaded
    // once per k-step, shared by both accumulators. Fold (1/denom)*sqrt(P).
    {
        const float* vb = pval + (size_t)h * P_ * D_;
        const int ct  = w & 3;
        const int rtb = (w >> 2) * 2;              // rt pair {rtb, rtb+1}
        const float* sr0 = S + (size_t)(rtb * 16 + ln) * STRIDE;
        const float* sr1 = sr0 + (size_t)16 * STRIDE;
        v8f c0 = {}, c1 = {};
        for (int kk = 0; kk < 32; ++kk) {
            const int kb = kk * 32;
            v16bf a0, a1, b;
            #pragma unroll
            for (int j = 0; j < 16; ++j) {
                const int ka = kb + kA(j, lh);
                a0[j] = f2b(sr0[ka]);                              // A from LDS probs
                a1[j] = f2b(sr1[ka]);
                const int p = kb + kA(j, lh);                      // B 32x16 from V
                b[j] = f2b(vb[(size_t)p * D_ + ct * 16 + ln]);
            }
            c0 = __builtin_amdgcn_wmma_f32_16x16x32_bf16(false, a0, false, b, (short)0, c0, false, false);
            c1 = __builtin_amdgcn_wmma_f32_16x16x32_bf16(false, a1, false, b, (short)0, c1, false, false);
        }
        #pragma unroll
        for (int r = 0; r < 8; ++r) {
            const int lr0 = rtb * 16 + lh * 8 + r;                 // local rows
            const int lr1 = lr0 + 16;
            const float v0 = c0[r] * invd[lr0] * 32.0f;            // sqrt(P)=32, /denom
            const float v1 = c1[r] * invd[lr1] * 32.0f;
            __builtin_nontemporal_store(v0, out + ((size_t)bh * M_ + (m0 + lr0)) * D_ + ct * 16 + ln);
            __builtin_nontemporal_store(v1, out + ((size_t)bh * M_ + (m0 + lr1)) * D_ + ct * 16 + ln);
        }
    }
}

extern "C" void kernel_launch(void* const* d_in, const int* in_sizes, int n_in,
                              void* d_out, int out_size, void* d_ws, size_t ws_size,
                              hipStream_t stream)
{
    (void)in_sizes; (void)n_in; (void)out_size; (void)d_ws; (void)ws_size;
    const float* q    = (const float*)d_in[0];
    const float* attn = (const float*)d_in[1];
    const float* pkey = (const float*)d_in[2];
    const float* pval = (const float*)d_in[3];
    float* attn_out = (float*)d_out;
    float* out      = attn_out + (size_t)BH_ * M_ * SPAN_;

    dim3 grid(M_ / MTILE, BH_);   // 16 x 32 = 512 workgroups
    dim3 block(256);              // 8 wave32
    const size_t lds_bytes = (size_t)(MTILE * STRIDE + MTILE) * sizeof(float); // ~257 KB of 320 KB WGP LDS
    hipLaunchKernelGGL(pers_mem_fused, grid, block, lds_bytes, stream,
                       q, attn, pkey, pval, attn_out, out);
}
